// GlobalMultiHeadAttention_62612033241269
// MI455X (gfx1250) — matrix operand order repaired and checked by reading, not compiled, not run on hardware
//
#include <hip/hip_runtime.h>
#include <math.h>

// ---------------------------------------------------------------------------
// Shapes (compile-time, from the reference): B=8, N=64, T=256, D=256, H=8
// BN = 512 groups. Per group (softmax over singleton key dim == 1, so the
// Q/K projections, bias-MLP and mask are mathematically dead):
//   out[t,:] = x[t,:] + v[:] + ( gelu(feat[t]@Vw1+Vb1) @ Vw2 + Vb2 )
// ---------------------------------------------------------------------------

#define TT 256          // tokens per group
#define DD 256          // model dim
#define HID 128         // value-MLP hidden
#define SH 136          // padded LDS row stride (halves) = 128 + 8
#define BN_TOTAL 512

typedef __attribute__((ext_vector_type(16))) _Float16 v16h;
typedef __attribute__((ext_vector_type(2)))  _Float16 v2h;
typedef __attribute__((ext_vector_type(8)))  float    v8f;
typedef __attribute__((ext_vector_type(4)))  float    v4f;

// exact GELU (matches jax.nn.gelu approximate=False)
__device__ __forceinline__ float gelu_exact(float x) {
    return 0.5f * x * (1.0f + erff(x * 0.70710678118654752f));
}

__device__ __forceinline__ v16h load_frag16(const _Float16* p0, const _Float16* p1) {
    v16h r;
    reinterpret_cast<uint4*>(&r)[0] = *reinterpret_cast<const uint4*>(p0);
    reinterpret_cast<uint4*>(&r)[1] = *reinterpret_cast<const uint4*>(p1);
    return r;
}

// ---------------------------------------------------------------------------
// Kernel 0: W2T[n][k] = (f16) Vw2[k][n]   (Vw2 is [128][256] row-major f32)
// ---------------------------------------------------------------------------
__global__ void k_prep_w2t(const float* __restrict__ Vw2, _Float16* __restrict__ w2t) {
    int idx = blockIdx.x * blockDim.x + threadIdx.x;   // 0 .. 32767
    if (idx < DD * HID) {
        int n = idx >> 7;          // 0..255
        int k = idx & 127;         // 0..127
        w2t[n * HID + k] = (_Float16)Vw2[k * DD + n];
    }
}

// ---------------------------------------------------------------------------
// Kernel 1: per (b,n) group -> masked summary, pos summary, v = summary@Wv+bv
// grid = 512 blocks x 256 threads
// ---------------------------------------------------------------------------
__global__ void __launch_bounds__(256)
k_summary(const float* __restrict__ x,
          const int*   __restrict__ masks,
          const float* __restrict__ pos,
          const float* __restrict__ Wv,
          const float* __restrict__ bv,
          float* __restrict__ v_ws,     // [BN][256]
          float* __restrict__ ps_ws) {  // [BN][4]
    __shared__ float ssum[DD];
    __shared__ int   smask[TT];
    __shared__ v4f   red[TT];     // tree reduction for pos sums + count

    const int bn  = blockIdx.x;
    const int tid = threadIdx.x;
    const float* xb = x + (size_t)bn * TT * DD;
    const float* pb = pos + (size_t)bn * TT * 3;

    const int m = masks[bn * TT + tid];
    smask[tid] = m;
    {   // per-token contribution (m*px, m*py, m*pz, m)
        const float fm = (float)m;
        v4f c;
        c.x = fm * pb[tid * 3 + 0];
        c.y = fm * pb[tid * 3 + 1];
        c.z = fm * pb[tid * 3 + 2];
        c.w = fm;
        red[tid] = c;
    }
    __syncthreads();

    // masked column sum over tokens (coalesced over d = tid)
    float acc = 0.0f;
    #pragma unroll 8
    for (int t = 0; t < TT; ++t)
        acc += (float)smask[t] * xb[t * DD + tid];

    // tree-reduce pos sums + count
    #pragma unroll
    for (int s = TT / 2; s > 0; s >>= 1) {
        if (tid < s) {
            v4f a = red[tid], b = red[tid + s];
            a.x += b.x; a.y += b.y; a.z += b.z; a.w += b.w;
            red[tid] = a;
        }
        __syncthreads();
    }

    const v4f tot = red[0];
    const float denom = fmaxf(tot.w, 1e-6f);
    ssum[tid] = acc / denom;
    if (tid == 0) {
        ps_ws[bn * 4 + 0] = tot.x / denom;
        ps_ws[bn * 4 + 1] = tot.y / denom;
        ps_ws[bn * 4 + 2] = tot.z / denom;
        ps_ws[bn * 4 + 3] = denom;
    }
    __syncthreads();

    // v[d] = bv[d] + sum_j summary[j] * Wv[j][d]
    float vacc = bv[tid];
    #pragma unroll 8
    for (int j = 0; j < DD; ++j)
        vacc = fmaf(ssum[j], Wv[j * DD + tid], vacc);
    v_ws[bn * DD + tid] = vacc;
}

// ---------------------------------------------------------------------------
// Kernel 2: per (b,n) group
//   phase A: feat -> gelu(feat@Vw1+Vb1) as f16 A-matrix in LDS [256][128]
//            stage W2T f16 in LDS, vpb[d] = v[d] + Vb2[d]
//   phase B: WMMA f16 GEMM [256,128]@[128,256], C init = vpb, + residual x
//            Outer loop over this wave's 2 M-tiles (A held in 32 VGPRs);
//            per N-tile: 4 fully independent WMMAs (one per K fragment,
//            no accumulation chain -> no WMMA RAW hazards), add-tree merge.
// grid = 512 blocks x 256 threads (8 waves)
// ---------------------------------------------------------------------------
__global__ void __launch_bounds__(256)
k_main(const float* __restrict__ x,
       const float* __restrict__ pos,
       const float* __restrict__ Vw1,    // [5][128]
       const float* __restrict__ Vb1,    // [128]
       const float* __restrict__ Vb2,    // [256]
       const _Float16* __restrict__ w2t, // [256][128] f16
       const float* __restrict__ v_ws,
       const float* __restrict__ ps_ws,
       float* __restrict__ out) {
    extern __shared__ char smem_raw[];
    _Float16* h1s = reinterpret_cast<_Float16*>(smem_raw);        // 256*136 halves
    _Float16* w2s = h1s + TT * SH;                                // 256*136 halves
    float*    vpb = reinterpret_cast<float*>(w2s + DD * SH);      // 256
    float*    w1s = vpb + DD;                                     // 5*128
    float*    b1s = w1s + 5 * HID;                                // 128
    float*    pss = b1s + HID;                                    // 4

    const int bn  = blockIdx.x;
    const int tid = threadIdx.x;

    // --- stage small tensors ---
    for (int i = tid; i < 5 * HID; i += TT) w1s[i] = Vw1[i];
    if (tid < HID) b1s[tid] = Vb1[tid];
    vpb[tid] = v_ws[bn * DD + tid] + Vb2[tid];
    if (tid < 4) pss[tid] = ps_ws[bn * 4 + tid];

    // --- stage W2T (64 KB) into padded LDS, 16B vectors ---
    {
        const uint4* wg = reinterpret_cast<const uint4*>(w2t);    // 4096 uint4
        #pragma unroll 4
        for (int i = tid; i < 4096; i += TT) {
            int r = i >> 4;        // row (0..255), 16 uint4 per row
            int c = i & 15;
            *reinterpret_cast<uint4*>(&w2s[r * SH + c * 8]) = wg[i];
        }
    }
    __syncthreads();   // pss/w1s/b1s visible

    // --- layer 1 MLP: token t = tid produces 128 gelu'd f16 activations ---
    {
        const float* pb = pos + (size_t)bn * TT * 3;
        const float f0 = pb[tid * 3 + 0] - pss[0];
        const float f1 = pb[tid * 3 + 1] - pss[1];
        const float f2 = pb[tid * 3 + 2] - pss[2];
        const float f4 = f0 * f0 + f1 * f1 + f2 * f2;   // dist^2
        const float f3 = sqrtf(f4);                     // dist
        _Float16* hrow = &h1s[tid * SH];
        #pragma unroll 4
        for (int j = 0; j < HID; j += 2) {
            float s0 = b1s[j];
            float s1 = b1s[j + 1];
            s0 = fmaf(f0, w1s[0 * HID + j], s0);  s1 = fmaf(f0, w1s[0 * HID + j + 1], s1);
            s0 = fmaf(f1, w1s[1 * HID + j], s0);  s1 = fmaf(f1, w1s[1 * HID + j + 1], s1);
            s0 = fmaf(f2, w1s[2 * HID + j], s0);  s1 = fmaf(f2, w1s[2 * HID + j + 1], s1);
            s0 = fmaf(f3, w1s[3 * HID + j], s0);  s1 = fmaf(f3, w1s[3 * HID + j + 1], s1);
            s0 = fmaf(f4, w1s[4 * HID + j], s0);  s1 = fmaf(f4, w1s[4 * HID + j + 1], s1);
            v2h p; p.x = (_Float16)gelu_exact(s0); p.y = (_Float16)gelu_exact(s1);
            *reinterpret_cast<v2h*>(&hrow[j]) = p;
        }
    }
    __syncthreads();   // h1s / w2s ready

    // --- WMMA GEMM: D = A[256,128] @ B[128,256] + C(vpb), then +x residual ---
    const int lane = tid & 31;
    const int wid  = tid >> 5;           // 0..7 -> M-tiles {2w, 2w+1}
    const int lm   = lane & 15;
    const int hi   = lane >> 4;          // 0 or 1

    // A fragment (ISA 16-bit A 16x32): lane<16 -> K{0..7,16..23}, lane>=16 -> K{8..15,24..31}
    const int aK = hi * 8;
    // B fragment (same K order as A): lane<16 -> K{0..7,16..23}, lane>=16 -> K{8..15,24..31}
    const int bK = hi * 8;

    const float* xg = x   + (size_t)bn * TT * DD;
    float*       og = out + (size_t)bn * TT * DD;

    for (int mm = 0; mm < 2; ++mm) {
        const int mrow  = wid * 32 + mm * 16;   // M-tile base row
        const int rbase = mrow + hi * 8;        // first row this lane owns in C/D

        // hold A fragments for this M-tile only (32 VGPRs)
        v16h Am[4];
        #pragma unroll
        for (int kt = 0; kt < 4; ++kt) {
            const _Float16* p = &h1s[(mrow + lm) * SH + kt * 32 + aK];
            Am[kt] = load_frag16(p, p + 16);
        }

        for (int nt = 0; nt < 16; ++nt) {
            const int col = nt * 16 + lm;

            // residual prefetch: 8 independent global loads, one wait,
            // hidden under the ds loads + WMMA group below
            float xr[8];
            #pragma unroll
            for (int r = 0; r < 8; ++r) xr[r] = xg[(rbase + r) * DD + col];

            v16h Bf[4];
            #pragma unroll
            for (int kt = 0; kt < 4; ++kt) {
                const _Float16* p = &w2s[(nt * 16 + lm) * SH + kt * 32 + bK];
                Bf[kt] = load_frag16(p, p + 16);
            }

            const float ci = vpb[col];
            v8f c0 = {ci, ci, ci, ci, ci, ci, ci, ci};
            v8f c1 = {0.f, 0.f, 0.f, 0.f, 0.f, 0.f, 0.f, 0.f};
            v8f c2 = c1;
            v8f c3 = c1;

            // 4 fully independent WMMAs (one per K fragment) - no RAW chain
            c0 = __builtin_amdgcn_wmma_f32_16x16x32_f16(false, Am[0], false, Bf[0], (short)0, c0, false, false);
            c1 = __builtin_amdgcn_wmma_f32_16x16x32_f16(false, Am[1], false, Bf[1], (short)0, c1, false, false);
            c2 = __builtin_amdgcn_wmma_f32_16x16x32_f16(false, Am[2], false, Bf[2], (short)0, c2, false, false);
            c3 = __builtin_amdgcn_wmma_f32_16x16x32_f16(false, Am[3], false, Bf[3], (short)0, c3, false, false);

            // C/D layout: lane<16 -> N=lane, M=r ; lane>=16 -> N=lane-16, M=r+8
            #pragma unroll
            for (int r = 0; r < 8; ++r) {
                og[(rbase + r) * DD + col] = xr[r] + ((c0[r] + c1[r]) + (c2[r] + c3[r]));
            }
        }
    }
}

// ---------------------------------------------------------------------------
// launch
// ---------------------------------------------------------------------------
extern "C" void kernel_launch(void* const* d_in, const int* in_sizes, int n_in,
                              void* d_out, int out_size, void* d_ws, size_t ws_size,
                              hipStream_t stream) {
    const float* x    = (const float*)d_in[0];
    const int*   mask = (const int*)  d_in[2];
    const float* pos  = (const float*)d_in[3];
    const float* Wv   = (const float*)d_in[8];
    const float* bv   = (const float*)d_in[9];
    const float* Vw1  = (const float*)d_in[14];
    const float* Vb1  = (const float*)d_in[15];
    const float* Vw2  = (const float*)d_in[16];
    const float* Vb2  = (const float*)d_in[17];
    float* out = (float*)d_out;

    // workspace layout
    char* ws = (char*)d_ws;
    _Float16* w2t   = (_Float16*)ws;                       // 256*128 f16 = 64 KB
    float*    v_ws  = (float*)(ws + 65536);                // 512*256 f32 = 512 KB
    float*    ps_ws = (float*)(ws + 65536 + 524288);       // 512*4 f32

    k_prep_w2t<<<(DD * HID + 255) / 256, 256, 0, stream>>>(Vw2, w2t);
    k_summary<<<BN_TOTAL, 256, 0, stream>>>(x, mask, pos, Wv, bv, v_ws, ps_ws);

    const int smem = (TT * SH + DD * SH) * (int)sizeof(_Float16)    // h1s + w2s
                   + (DD + 5 * HID + HID + 4) * (int)sizeof(float); // vpb/w1s/b1s/pss
    k_main<<<BN_TOTAL, 256, smem, stream>>>(x, pos, Vw1, Vb1, Vb2, w2t, v_ws, ps_ws, out);
}
